// CausalSelfAttention_20933670601118
// MI455X (gfx1250) — matrix operand order repaired and checked by reading, not compiled, not run on hardware
//
#include <hip/hip_runtime.h>

typedef __attribute__((ext_vector_type(16))) _Float16 v16h;
typedef __attribute__((ext_vector_type(8)))  float    v8f;

#define D_MODEL 1024
#define SEQ     2048
#define NHEADS  16
#define HDIM    64
#define BATCH   2
#define NROWS   (BATCH * SEQ)   // 4096

union U16 { v16h v; unsigned u[8]; _Float16 h[16]; };

__device__ __forceinline__ v8f wmma_f16(v16h a, v16h b, v8f c) {
  // 8 args: (neg_a, A, neg_b, B, c_mod, C, reuse_a, reuse_b)
  return __builtin_amdgcn_wmma_f32_16x16x32_f16(false, a, false, b, (short)0, c,
                                                false, false);
}

// WGP-scope prefetch (scope field 0 -> pulls into all cache levels incl. WGP$).
// __builtin_prefetch can only reach SE/SYS scope, which stops at GL2.
__device__ __forceinline__ void prefetch_wgp(const void* p) {
  asm volatile("global_prefetch_b8 %0, off" :: "v"(p));
}

__global__ void __launch_bounds__(256) cvt_f32_f16(const float* __restrict__ in,
                                                   _Float16* __restrict__ out, int n) {
  int i = blockIdx.x * 256 + threadIdx.x;
  if (i < n) out[i] = (_Float16)in[i];
}

// C[16x64 per wave] = A[M,K] * Bw[N,K]^T  with K = D_MODEL.
// mode 0: RoPE + 0.125 scale -> half, [b,h,s,d]   (Q)
// mode 1: RoPE               -> half, [b,h,s,d]   (K)
// mode 2: plain              -> half, [b,h,d,s]   (V, TRANSPOSED for PV B-tiles)
// mode 3: plain              -> float, [row, col] (final projection)
__global__ void __launch_bounds__(128) gemm_head_kernel(
    const _Float16* __restrict__ A, const _Float16* __restrict__ Bw,
    const float* __restrict__ cosT, const float* __restrict__ sinT,
    _Float16* __restrict__ dstH, float* __restrict__ dstF, int mode) {
  const int lane = threadIdx.x & 31;
  const int hi = lane >> 4;        // lane half (0/1)
  const int ln = lane & 15;
  const int wave = blockIdx.x * 4 + (threadIdx.x >> 5);  // 0..4095
  const int rowTile = wave >> 4;   // 0..255  (M / 16)
  const int head = wave & 15;      // N quad of 64 = head
  const int rowBase = rowTile << 4;

  v8f acc[4] = {};
  const unsigned* arow =
      (const unsigned*)(A + (size_t)(rowBase + ln) * D_MODEL);
  const unsigned* brow0 = (const unsigned*)(
      Bw + (size_t)(head * 64 + ln) * D_MODEL);   // B tile t=0 row base

  for (int k0 = 0; k0 < D_MODEL; k0 += 32) {
    // WGP-scope prefetch 8 K-steps ahead, branch-free: the lookahead wraps
    // within the current row so the (non-speculative) prefetch address is
    // always valid and the loop body stays a single basic block.
    {
      const int pk = ((k0 + 256) & (D_MODEL - 1)) >> 1;
      prefetch_wgp((const void*)(arow + pk));
      prefetch_wgp((const void*)(brow0 + pk));
    }
    // A 16x32 f16, A-layout: lane = M; half j -> K = j + 16*hi
    U16 a;
#pragma unroll
    for (int j2 = 0; j2 < 8; ++j2) {
      int kk = 2 * j2 + 16 * hi + k0;
      a.u[j2] = arow[kk >> 1];
    }
#pragma unroll
    for (int t = 0; t < 4; ++t) {
      // B 32x16 f16, B-layout: lane = N; half j -> K = j + 16*hi
      // B(k,n) = Bw[n, k]  (x @ W^T)
      U16 b;
      const unsigned* brow = (const unsigned*)(
          Bw + (size_t)(head * 64 + t * 16 + ln) * D_MODEL + k0 + hi * 16);
#pragma unroll
      for (int j2 = 0; j2 < 8; ++j2) b.u[j2] = brow[j2];
      acc[t] = wmma_f16(a.v, b.v, acc[t]);
    }
  }

  // Epilogue. C layout: VGPR r, lane -> (m = r + 8*hi, n = ln)
#pragma unroll
  for (int t = 0; t < 4; ++t) {
#pragma unroll
    for (int r = 0; r < 8; ++r) {
      const int m = r + 8 * hi;
      const int gm = rowBase + m;          // row in [0, B*S)
      const int bb = gm >> 11;             // / SEQ
      const int s = gm & (SEQ - 1);
      const int d = t * 16 + ln;           // dim within head, [0,64)
      float val = acc[t][r];
      if (mode <= 1) {
        // rotate_half: d<32 pairs with d+32 (tile t+2), d>=32 with d-32 (t-2)
        const float rot = (t < 2) ? -acc[t + 2][r] : acc[t - 2][r];
        const float cc = cosT[s * HDIM + d];
        const float ss = sinT[s * HDIM + d];
        val = val * cc + rot * ss;
        if (mode == 0) val *= 0.125f;      // fold 1/sqrt(HDIM) into Q
        dstH[(((size_t)bb * NHEADS + head) * SEQ + s) * HDIM + d] =
            (_Float16)val;
      } else if (mode == 2) {
        // V transposed: [b, h, d, s] so PV B-tiles read contiguous keys
        dstH[(((size_t)bb * NHEADS + head) * HDIM + d) * SEQ + s] =
            (_Float16)val;
      } else {  // mode 3
        dstF[(size_t)gm * D_MODEL + head * HDIM + d] = val;
      }
    }
  }
}

// Flash attention: one wave per (b, h, 16-query tile). Causal.
// Vt is the transposed V: [b, h, d, s].
__global__ void __launch_bounds__(128) attn_kernel(
    const _Float16* __restrict__ Qh, const _Float16* __restrict__ Kh,
    const _Float16* __restrict__ Vt, _Float16* __restrict__ Oc) {
  __shared__ _Float16 Pl[4][16][32];   // per-wave P staging (C->A relayout)
  const int lane = threadIdx.x & 31;
  const int wsl = threadIdx.x >> 5;
  const int hi = lane >> 4;
  const int ln = lane & 15;
  const int wave = blockIdx.x * 4 + wsl;        // 0..4095
  const int qTile = wave & (SEQ / 16 - 1);      // 0..127
  const int bh = wave >> 7;                     // 0..31
  const int b = bh >> 4, h = bh & 15;
  const size_t base = (size_t)bh * SEQ * HDIM;  // for Qh/Kh [b,h,s,d]
  const int s0 = qTile << 4;

  // Preload Q (16 queries x 64 dims) as two A-matrices (dims 0-31, 32-63)
  U16 aQ0, aQ1;
  {
    const unsigned* qrow =
        (const unsigned*)(Qh + base + (size_t)(s0 + ln) * HDIM);
#pragma unroll
    for (int j2 = 0; j2 < 8; ++j2) {
      int kk = 2 * j2 + 16 * hi;
      aQ0.u[j2] = qrow[kk >> 1];
      aQ1.u[j2] = qrow[(kk + 32) >> 1];
    }
  }

  v8f o[4] = {};
  float rm[8], rl[8];
#pragma unroll
  for (int r = 0; r < 8; ++r) { rm[r] = -3.0e38f; rl[r] = 0.0f; }

  const int nk = s0 + 16;                 // causal: keys [0, nk)
  const int chunks = (nk + 31) >> 5;      // 32-key chunks
  for (int c = 0; c < chunks; ++c) {
    const int kb = c << 5;
    // scores: S = Q * K^T  (two 16-key tiles, K-dim = 64 -> 2 wmma steps each)
    v8f sc[2] = {};
#pragma unroll
    for (int t = 0; t < 2; ++t) {
      const unsigned* krow = (const unsigned*)(
          Kh + base + (size_t)(kb + t * 16 + ln) * HDIM);
      U16 bk0, bk1;
#pragma unroll
      for (int j2 = 0; j2 < 8; ++j2) {
        bk0.u[j2] = krow[(hi * 16) / 2 + j2];          // dims 0-31
        bk1.u[j2] = krow[(32 + hi * 16) / 2 + j2];     // dims 32-63
      }
      sc[t] = wmma_f16(aQ0.v, bk0.v, sc[t]);
      sc[t] = wmma_f16(aQ1.v, bk1.v, sc[t]);
    }
    // causal mask + online softmax (per-row reductions within 16-lane halves)
#pragma unroll
    for (int r = 0; r < 8; ++r) {
      const int q = s0 + r + 8 * hi;
      if (kb + ln > q)      sc[0][r] = -1.0e9f;
      if (kb + 16 + ln > q) sc[1][r] = -1.0e9f;
      float mx = fmaxf(sc[0][r], sc[1][r]);
      mx = fmaxf(mx, __shfl_xor(mx, 1, 32));
      mx = fmaxf(mx, __shfl_xor(mx, 2, 32));
      mx = fmaxf(mx, __shfl_xor(mx, 4, 32));
      mx = fmaxf(mx, __shfl_xor(mx, 8, 32));
      const float mn = fmaxf(rm[r], mx);
      const float al = __expf(rm[r] - mn);
      const float p0 = __expf(sc[0][r] - mn);
      const float p1 = __expf(sc[1][r] - mn);
      float sm = p0 + p1;
      sm += __shfl_xor(sm, 1, 32);
      sm += __shfl_xor(sm, 2, 32);
      sm += __shfl_xor(sm, 4, 32);
      sm += __shfl_xor(sm, 8, 32);
      rl[r] = rl[r] * al + sm;
      rm[r] = mn;
      sc[0][r] = p0;
      sc[1][r] = p1;
#pragma unroll
      for (int t = 0; t < 4; ++t) o[t][r] *= al;
    }
    // P: C-layout -> LDS -> A-layout (16 x 32)
#pragma unroll
    for (int t = 0; t < 2; ++t)
#pragma unroll
      for (int r = 0; r < 8; ++r)
        Pl[wsl][r + 8 * hi][t * 16 + ln] = (_Float16)sc[t][r];
    asm volatile("s_wait_dscnt 0" ::: "memory");
    U16 aP;
#pragma unroll
    for (int j = 0; j < 16; ++j) {
      const int kk = j + 16 * hi;
      aP.h[j] = Pl[wsl][ln][kk];
    }
    // O += P * V : B(k=key, n=dim) from transposed V -> contiguous per lane
#pragma unroll
    for (int t = 0; t < 4; ++t) {
      U16 bv;
      const unsigned* vtrow = (const unsigned*)(
          Vt + ((size_t)bh * HDIM + t * 16 + ln) * SEQ + kb + 16 * hi);
#pragma unroll
      for (int j2 = 0; j2 < 8; ++j2) bv.u[j2] = vtrow[j2];
      o[t] = wmma_f16(aP.v, bv.v, o[t]);
    }
  }
  // finalize: O / l, store fp16 into [B*S, D] for the output projection
#pragma unroll
  for (int t = 0; t < 4; ++t)
#pragma unroll
    for (int r = 0; r < 8; ++r) {
      const int s = s0 + r + 8 * hi;
      const int d = t * 16 + ln;
      Oc[((size_t)b * SEQ + s) * D_MODEL + h * HDIM + d] =
          (_Float16)(o[t][r] / rl[r]);
    }
}

extern "C" void kernel_launch(void* const* d_in, const int* in_sizes, int n_in,
                              void* d_out, int out_size, void* d_ws, size_t ws_size,
                              hipStream_t stream) {
  const float* x    = (const float*)d_in[0];
  const float* cosT = (const float*)d_in[1];
  const float* sinT = (const float*)d_in[2];
  const float* Wq   = (const float*)d_in[3];
  const float* Wk   = (const float*)d_in[4];
  const float* Wv   = (const float*)d_in[5];
  const float* Wo   = (const float*)d_in[6];
  float* out = (float*)d_out;

  const size_t NX = (size_t)NROWS * D_MODEL;    // 4,194,304
  const size_t NW = (size_t)D_MODEL * D_MODEL;  // 1,048,576
  _Float16* wsp = (_Float16*)d_ws;
  _Float16* Xh  = wsp;
  _Float16* Wqh = Xh + NX;
  _Float16* Wkh = Wqh + NW;
  _Float16* Wvh = Wkh + NW;
  _Float16* Woh = Wvh + NW;
  _Float16* Qh  = Woh + NW;
  _Float16* Kh  = Qh + NX;
  _Float16* Vt  = Kh + NX;   // V transposed, [b,h,d,s]
  _Float16* Ah  = Vt + NX;   // attention output, [B*S, D] fp16

  cvt_f32_f16<<<(int)(NX / 256), 256, 0, stream>>>(x, Xh, (int)NX);
  cvt_f32_f16<<<(int)(NW / 256), 256, 0, stream>>>(Wq, Wqh, (int)NW);
  cvt_f32_f16<<<(int)(NW / 256), 256, 0, stream>>>(Wk, Wkh, (int)NW);
  cvt_f32_f16<<<(int)(NW / 256), 256, 0, stream>>>(Wv, Wvh, (int)NW);
  cvt_f32_f16<<<(int)(NW / 256), 256, 0, stream>>>(Wo, Woh, (int)NW);

  // 4096 waves: 256 row-tiles x 16 heads; 4 waves / 128-thread block
  gemm_head_kernel<<<1024, 128, 0, stream>>>(Xh, Wqh, cosT, sinT, Qh, nullptr, 0);
  gemm_head_kernel<<<1024, 128, 0, stream>>>(Xh, Wkh, cosT, sinT, Kh, nullptr, 1);
  gemm_head_kernel<<<1024, 128, 0, stream>>>(Xh, Wvh, cosT, sinT, Vt, nullptr, 2);

  // 4096 waves: 2 batch x 16 heads x 128 query tiles
  attn_kernel<<<1024, 128, 0, stream>>>(Qh, Kh, Vt, Ah);

  // final projection: out = Ah @ Wo^T, fp32 store
  gemm_head_kernel<<<1024, 128, 0, stream>>>(Ah, Woh, cosT, sinT, nullptr, out, 3);
}
